// TrajectoryTensorGRU_42494406426980
// MI455X (gfx1250) — matrix operand order repaired and checked by reading, not compiled, not run on hardware
//
#include <hip/hip_runtime.h>
#include <math.h>

typedef __attribute__((ext_vector_type(16))) __bf16 v16bf;
typedef __attribute__((ext_vector_type(8)))  float  v8f;

#define BB 2048   // batch
#define II 256    // input features
#define TT 128    // time steps
#define HH 512    // hidden
#define GG 2048   // packed gate output cols: [r | z | i_n | h_n]
#define KK 768    // packed K: [x (256) | h (512)]

static __device__ __forceinline__ unsigned short f2bf(float f) {
    union { float f; unsigned u; } v; v.f = f;
    unsigned r = v.u + 0x7FFFu + ((v.u >> 16) & 1u);   // round-to-nearest-even
    return (unsigned short)(r >> 16);
}

// ---------------------------------------------------------------------------
// Pack Wbig [2048 x 768] bf16 row-major.
// rows   0..511  : [Wih_r | Whh_r]
// rows 512..1023 : [Wih_z | Whh_z]
// rows 1024..1535: [Wih_n |   0  ]   (i_n)
// rows 1536..2047: [  0   | Whh_n]   (h_n)
// ---------------------------------------------------------------------------
__global__ void pack_weights(const float* __restrict__ wih,
                             const float* __restrict__ whh,
                             unsigned short* __restrict__ wbig) {
    int idx = blockIdx.x * 256 + threadIdx.x;
    if (idx >= GG * KK) return;
    int o = idx / KK;
    int k = idx % KK;
    float v;
    if (o < 1024)      v = (k < 256) ? wih[o * II + k] : whh[o * HH + (k - 256)];
    else if (o < 1536) v = (k < 256) ? wih[o * II + k] : 0.0f;
    else               v = (k < 256) ? 0.0f : whh[(o - 512) * HH + (k - 256)];
    wbig[idx] = f2bf(v);
}

// ---------------------------------------------------------------------------
// Transpose+convert input [B*I, T] f32 -> xT [T, B*I] bf16 (tiled via LDS).
// ---------------------------------------------------------------------------
__global__ void transpose_input(const float* __restrict__ in,
                                unsigned short* __restrict__ xT) {
    __shared__ float tile[32][33];
    const int R = BB * II;                 // 524288 rows, C = TT = 128 cols
    int c0 = blockIdx.x * 32;              // along T
    int r0 = blockIdx.y * 32;              // along B*I
    int tx = threadIdx.x, ty = threadIdx.y;
    #pragma unroll
    for (int k = 0; k < 4; ++k) {
        int rr = r0 + ty + k * 8;
        tile[ty + k * 8][tx] = in[(size_t)rr * TT + c0 + tx];
    }
    __syncthreads();
    #pragma unroll
    for (int k = 0; k < 4; ++k) {
        int cc = c0 + ty + k * 8;
        xT[(size_t)cc * R + r0 + tx] = f2bf(tile[tx][ty + k * 8]);
    }
}

__global__ void zero_h(float* __restrict__ hf, unsigned short* __restrict__ hb) {
    int idx = blockIdx.x * 256 + threadIdx.x;
    if (idx < BB * HH) { hf[idx] = 0.0f; hb[idx] = 0; }
}

// ---------------------------------------------------------------------------
// Per-step GEMM: g[B, 2048] = A[B, 768] x Wbig^T, bf16 WMMA, f32 accum.
// A cols 0..255 come from xT (current timestep), cols 256..767 from h_bf16.
// Grid (16, 32): block tile 128(M) x 64(N), 4 waves, wave tile 32 x 64.
// Per-N-tile K range skips the structurally-zero weight blocks.
// ---------------------------------------------------------------------------
__global__ __launch_bounds__(128)
void gru_step_gemm(const unsigned short* __restrict__ xT_t,   // [B, 256] bf16
                   const unsigned short* __restrict__ hb,     // [B, 512] bf16
                   const unsigned short* __restrict__ wbig,   // [2048, 768] bf16
                   float* __restrict__ g) {                   // [B, 2048] f32
    const int wave = threadIdx.x >> 5;
    const int lane = threadIdx.x & 31;
    const int lmod = lane & 15;
    const int lhalf = lane >> 4;           // 0: lanes 0-15, 1: lanes 16-31

    const int m0 = blockIdx.x * 128 + wave * 32;   // wave's 32 rows
    const int n0 = blockIdx.y * 64;                // block's 64 cols

    int kbeg, kend;
    if (n0 < 1024)      { kbeg = 0;   kend = 768; }   // r, z combined
    else if (n0 < 1536) { kbeg = 0;   kend = 256; }   // i_n: input only
    else                { kbeg = 256; kend = 768; }   // h_n: hidden only

    v8f acc[2][4];
    #pragma unroll
    for (int i = 0; i < 2; ++i)
        #pragma unroll
        for (int j = 0; j < 4; ++j)
            acc[i][j] = (v8f){0.f, 0.f, 0.f, 0.f, 0.f, 0.f, 0.f, 0.f};

    union AF { uint4 q[2]; v16bf v; };

    for (int k0 = kbeg; k0 < kend; k0 += 32) {
        const unsigned short* aptr;
        int astride, ak;
        if (k0 < 256) { aptr = xT_t; astride = II; ak = k0; }
        else          { aptr = hb;   astride = HH; ak = k0 - 256; }

        // A fragments: lane lmod = row; half 0 holds K {0..7,16..23}, half 1 {8..15,24..31}
        AF afrag[2];
        #pragma unroll
        for (int mi = 0; mi < 2; ++mi) {
            size_t row = (size_t)(m0 + mi * 16 + lmod);
            size_t o0 = row * astride + ak + lhalf * 8;
            afrag[mi].q[0] = *(const uint4*)(aptr + o0);
            afrag[mi].q[1] = *(const uint4*)(aptr + o0 + 16);
        }
        // B fragments: column n = n0+nf*16+lmod is row of Wbig (contiguous K);
        // half 0 holds K {0..7,16..23}, half 1 {8..15,24..31}.
        v16bf bfrag[4];
        #pragma unroll
        for (int nf = 0; nf < 4; ++nf) {
            size_t row = (size_t)(n0 + nf * 16 + lmod);
            AF bt;
            size_t ob = row * KK + k0 + lhalf * 8;
            bt.q[0] = *(const uint4*)(wbig + ob);
            bt.q[1] = *(const uint4*)(wbig + ob + 16);
            bfrag[nf] = bt.v;
        }
        #pragma unroll
        for (int mi = 0; mi < 2; ++mi)
            #pragma unroll
            for (int nf = 0; nf < 4; ++nf)
                acc[mi][nf] = __builtin_amdgcn_wmma_f32_16x16x32_bf16(
                    false, afrag[mi].v, false, bfrag[nf],
                    (short)0, acc[mi][nf], false, false);
    }

    // Store: C frag lane layout — n = lmod, rows m = mhalf*8 + r.
    #pragma unroll
    for (int mi = 0; mi < 2; ++mi)
        #pragma unroll
        for (int nf = 0; nf < 4; ++nf) {
            int col = n0 + nf * 16 + lmod;
            int mrow = m0 + mi * 16 + lhalf * 8;
            #pragma unroll
            for (int r = 0; r < 8; ++r)
                g[(size_t)(mrow + r) * GG + col] = acc[mi][nf][r];
        }
}

// ---------------------------------------------------------------------------
// Fused gate nonlinearity + state update (one thread per (b, j)).
// ---------------------------------------------------------------------------
__global__ void gru_step_gates(const float* __restrict__ g,
                               const float* __restrict__ bih,
                               const float* __restrict__ bhh,
                               float* __restrict__ hf,
                               unsigned short* __restrict__ hb,
                               float* __restrict__ out) {
    int idx = blockIdx.x * 256 + threadIdx.x;   // b*512 + j
    int b = idx >> 9;
    int j = idx & 511;
    const float* gr_row = g + (size_t)b * GG;
    float gr  = gr_row[j]         + bih[j]        + bhh[j];
    float gz  = gr_row[512 + j]   + bih[512 + j]  + bhh[512 + j];
    float gin = gr_row[1024 + j]  + bih[1024 + j];
    float ghn = gr_row[1536 + j]  + bhh[1024 + j];
    float r = 1.0f / (1.0f + __expf(-gr));
    float z = 1.0f / (1.0f + __expf(-gz));
    float n = tanhf(gin + r * ghn);
    float hp = hf[idx];
    float hn = (1.0f - z) * n + z * hp;
    hf[idx] = hn;
    hb[idx] = f2bf(hn);
    if (out) out[idx] = hn;
}

extern "C" void kernel_launch(void* const* d_in, const int* in_sizes, int n_in,
                              void* d_out, int out_size, void* d_ws, size_t ws_size,
                              hipStream_t stream) {
    const float* input = (const float*)d_in[0];   // [B, I, T]
    const float* wih   = (const float*)d_in[1];   // [3H, I]
    const float* whh   = (const float*)d_in[2];   // [3H, H]
    const float* bih   = (const float*)d_in[3];   // [3H]
    const float* bhh   = (const float*)d_in[4];   // [3H]
    float* out = (float*)d_out;                   // [B, H]

    // Workspace layout (all sizes 256B-multiples)
    char* p = (char*)d_ws;
    unsigned short* xT   = (unsigned short*)p; p += (size_t)TT * BB * II * 2; // 128 MB
    unsigned short* wbig = (unsigned short*)p; p += (size_t)GG * KK * 2;      //   3 MB
    float*          g    = (float*)p;          p += (size_t)BB * GG * 4;      //  16 MB
    float*          hf   = (float*)p;          p += (size_t)BB * HH * 4;      //   4 MB
    unsigned short* hb   = (unsigned short*)p;                                //   2 MB

    pack_weights<<<(GG * KK + 255) / 256, 256, 0, stream>>>(wih, whh, wbig);
    transpose_input<<<dim3(TT / 32, (BB * II) / 32), dim3(32, 8), 0, stream>>>(input, xT);
    zero_h<<<(BB * HH + 255) / 256, 256, 0, stream>>>(hf, hb);

    for (int t = 0; t < TT; ++t) {
        const unsigned short* xT_t = xT + (size_t)t * BB * II;
        gru_step_gemm<<<dim3(BB / 128, GG / 64), 128, 0, stream>>>(xT_t, hb, wbig, g);
        gru_step_gates<<<(BB * HH) / 256, 256, 0, stream>>>(
            g, bih, bhh, hf, hb, (t == TT - 1) ? out : nullptr);
    }
}
